// FA_44633300140431
// MI455X (gfx1250) — matrix operand order repaired and checked by reading, not compiled, not run on hardware
//
#include <hip/hip_runtime.h>
#include <hip/hip_bf16.h>

typedef __attribute__((ext_vector_type(16))) _Float16 v16h;
typedef __attribute__((ext_vector_type(8)))  _Float16 v8h;
typedef __attribute__((ext_vector_type(4)))  _Float16 v4h;
typedef __attribute__((ext_vector_type(8)))  float    v8f;

#define LROW   1000
#define NROWS  90112          // 4096 * 22
#define NMODES 64
#define KPAD   1024           // phase-A K padded to multiple of 32
#define XH_STRIDE 1032        // f16 x tile stride (bank-conflict free: 2064B -> +4 banks/row)
#define XS_STRIDE 1004        // f32 x tile stride (rows m and m+8 on disjoint banks)
#define CH_STRIDE 136         // f16 coeff tile stride (272B -> +4 banks/row)

#define TWO_PI_OVER_L 0.006283185307179587f   // 2*pi/1000
#define INV_SQRT_L    0.031622776601683794f   // 1/sqrt(1000)

// ---------------------------------------------------------------------------
// Setup kernel 1: analysis basis (phase A), weights folded in.
// Logical matrix: Ba[k][c], k=0..1023 (time), c=0..127 (0..63 = wr*cos, 64..127 = -wi*sin).
// Stored pre-swizzled as WMMA B-operand blocks: blk = ntile*32 + kstep,
// each block = 32 lanes x 16 f16; lane<16 -> col=ntile*16+lane, K=kstep*32+0..15;
// lane>=16 -> same col set, K=kstep*32+16..31 (contiguous per lane).
// ---------------------------------------------------------------------------
__global__ void fa_basisA(const float* __restrict__ fw, const float* __restrict__ fwi,
                          const int* __restrict__ index, _Float16* __restrict__ basisA) {
  const int blk   = blockIdx.x;        // 0..255
  const int lane  = threadIdx.x;       // 0..31
  const int ntile = blk >> 5;
  const int kstep = blk & 31;
  const int half  = lane >> 4;
  const int c     = ntile * 16 + (lane & 15);   // coefficient column 0..127
  v16h t;
#pragma unroll
  for (int j = 0; j < 16; ++j) {
    const int k = kstep * 32 + half * 16 + j;   // time index (B-matrix row)
    float v = 0.f;
    if (k < LROW) {
      if (c < NMODES) {
        const int widx = (c == 0) ? 0 : (c + 1);          // reference's widx quirk
        const int p = (index[c] * k) % LROW;              // exact phase reduction
        v = fw[widx] * cosf((float)p * TWO_PI_OVER_L) * INV_SQRT_L;
      } else {
        const int c2   = c - NMODES;
        const int widx = (c2 == 0) ? 0 : (c2 + 1);
        const int p = (index[c2] * k) % LROW;
        v = -fwi[widx] * sinf((float)p * TWO_PI_OVER_L) * INV_SQRT_L;
      }
    }
    t[j] = (_Float16)v;
  }
  ((v16h*)basisA)[blk * 32 + lane] = t;
}

// ---------------------------------------------------------------------------
// Setup kernel 2: synthesis basis (phase B).
// Logical matrix: Bb[k][n], k=0..127, n=0..1023 (cols >= 1000 never stored).
//   k in 0..63 :  s_k * cos(2*pi*k*n/L),  s_0 = 1/sqrt(L), s_k = 2/sqrt(L)
//   k in 64..127: k2=k-64; k2==0 -> 0 (irfft drops DC imag); else -2/sqrt(L)*sin(2*pi*k2*n/L)
// Same B-operand swizzle: blk = ntile*4 + kstep.
// ---------------------------------------------------------------------------
__global__ void fa_basisB(_Float16* __restrict__ basisB) {
  const int blk   = blockIdx.x;        // 0..255
  const int lane  = threadIdx.x;
  const int ntile = blk >> 2;          // 0..63
  const int kstep = blk & 3;           // 0..3
  const int half  = lane >> 4;
  const int n     = ntile * 16 + (lane & 15);   // output column 0..1023
  v16h t;
#pragma unroll
  for (int j = 0; j < 16; ++j) {
    const int k = kstep * 32 + half * 16 + j;   // coefficient index 0..127
    float v;
    if (k < NMODES) {
      const float s = (k == 0) ? INV_SQRT_L : (2.f * INV_SQRT_L);
      const int p = (k * n) % LROW;
      v = s * cosf((float)p * TWO_PI_OVER_L);
    } else {
      const int k2 = k - NMODES;
      if (k2 == 0) {
        v = 0.f;
      } else {
        const int p = (k2 * n) % LROW;
        v = -2.f * INV_SQRT_L * sinf((float)p * TWO_PI_OVER_L);
      }
    }
    t[j] = (_Float16)v;
  }
  ((v16h*)basisB)[blk * 32 + lane] = t;
}

// ---------------------------------------------------------------------------
// Fused main kernel: one workgroup (4 waves) per 16-row tile.
// ---------------------------------------------------------------------------
__global__ __launch_bounds__(128)
void fa_main(const float* __restrict__ x, const _Float16* __restrict__ basisA,
             const _Float16* __restrict__ basisB, float* __restrict__ out) {
  __shared__ float    xs[16 * XS_STRIDE];   // fp32 x tile (for exact residual add)
  __shared__ _Float16 xh[16 * XH_STRIDE];   // f16 x tile, K padded to 1024 with zeros
  __shared__ _Float16 ch[16 * CH_STRIDE];   // f16 coefficient tile (16 x 128)

  const int tid  = threadIdx.x;
  const int wave = tid >> 5;
  const int lane = tid & 31;
  const int half = lane >> 4;
  const int l16  = lane & 15;
  const long rowbase = (long)blockIdx.x * 16;

  // ---- stage x tile: 16 rows x 1000 f32 -> LDS (f32 + f16, zero-padded) ----
  for (int r = 0; r < 16; ++r) {
    const float4* src = (const float4*)(x + (rowbase + r) * LROW);
#pragma unroll
    for (int c4 = tid; c4 < 256; c4 += 128) {       // 256*4 = 1024 padded cols
      float4 v = make_float4(0.f, 0.f, 0.f, 0.f);
      if (c4 < 250) v = src[c4];                    // 250*4 = 1000 real cols
      if (c4 < 250) *(float4*)&xs[r * XS_STRIDE + c4 * 4] = v;
      v4h hv = { (_Float16)v.x, (_Float16)v.y, (_Float16)v.z, (_Float16)v.w };
      *(v4h*)&xh[r * XH_STRIDE + c4 * 4] = hv;
    }
  }
  __syncthreads();

  // ---- Phase A: coeff[16 x 128] = x_tile[16 x 1024] * basisA[1024 x 128] ----
  // Each wave owns 2 of the 8 coefficient N-tiles.
  const int nt0 = wave * 2;
  const int nt1 = wave * 2 + 1;
  v8f acc0 = {};
  v8f acc1 = {};
  const _Float16* xrow = &xh[l16 * XH_STRIDE + half * 16];  // A-frag contiguous: K+16*half+0..15
  const v16h* bA = (const v16h*)basisA;
#pragma unroll 4
  for (int ks = 0; ks < 32; ++ks) {
    v8h a0 = *(const v8h*)(xrow + ks * 32);
    v8h a1 = *(const v8h*)(xrow + ks * 32 + 8);
    v16h a = __builtin_shufflevector(a0, a1, 0,1,2,3,4,5,6,7,8,9,10,11,12,13,14,15);
    v16h b0 = bA[(nt0 * 32 + ks) * 32 + lane];
    v16h b1 = bA[(nt1 * 32 + ks) * 32 + lane];
    acc0 = __builtin_amdgcn_wmma_f32_16x16x32_f16(false, a, false, b0, (short)0, acc0, false, false);
    acc1 = __builtin_amdgcn_wmma_f32_16x16x32_f16(false, a, false, b1, (short)0, acc1, false, false);
  }

  // scatter accumulators into coefficient tile (row-major f16, padded stride)
#pragma unroll
  for (int v = 0; v < 8; ++v) {
    const int m = v + 8 * half;                 // C/D layout: VGPR v -> row v (+8 for hi lanes)
    ch[m * CH_STRIDE + nt0 * 16 + l16] = (_Float16)acc0[v];
    ch[m * CH_STRIDE + nt1 * 16 + l16] = (_Float16)acc1[v];
  }
  __syncthreads();

  // ---- Phase B: y[16 x 1000] = coeff[16 x 128] * basisB[128 x 1024] + x ----
  v16h afrag[4];
  const _Float16* crow = &ch[l16 * CH_STRIDE + half * 16];
#pragma unroll
  for (int ks = 0; ks < 4; ++ks) {
    v8h a0 = *(const v8h*)(crow + ks * 32);
    v8h a1 = *(const v8h*)(crow + ks * 32 + 8);
    afrag[ks] = __builtin_shufflevector(a0, a1, 0,1,2,3,4,5,6,7,8,9,10,11,12,13,14,15);
  }
  const v16h* bB = (const v16h*)basisB;
  for (int nt = wave; nt < 64; nt += 4) {       // 16 output N-tiles per wave
    v8f acc = {};
#pragma unroll
    for (int ks = 0; ks < 4; ++ks) {
      v16h b = bB[(nt * 4 + ks) * 32 + lane];
      acc = __builtin_amdgcn_wmma_f32_16x16x32_f16(false, afrag[ks], false, b, (short)0, acc, false, false);
    }
    const int col = nt * 16 + l16;
    if (col < LROW) {                           // masking only after all WMMAs (EXEC all-1s rule)
#pragma unroll
      for (int v = 0; v < 8; ++v) {
        const int m = v + 8 * half;
        out[(rowbase + m) * LROW + col] = acc[v] + xs[m * XS_STRIDE + col];
      }
    }
  }
}

extern "C" void kernel_launch(void* const* d_in, const int* in_sizes, int n_in,
                              void* d_out, int out_size, void* d_ws, size_t ws_size,
                              hipStream_t stream) {
  (void)in_sizes; (void)n_in; (void)out_size; (void)ws_size;
  const float* x     = (const float*)d_in[0];   // (4096,1,22,1000) f32
  const float* fw    = (const float*)d_in[1];   // (65,1) f32
  const float* fwi   = (const float*)d_in[2];   // (65,1) f32
  const int*   index = (const int*)d_in[3];     // (64,) i32

  _Float16* basisA = (_Float16*)d_ws;           // 256 blocks * 512 f16 = 256 KB
  _Float16* basisB = basisA + 256 * 512;        // another 256 KB

  fa_basisA<<<256, 32, 0, stream>>>(fw, fwi, index, basisA);
  fa_basisB<<<256, 32, 0, stream>>>(basisB);
  fa_main<<<NROWS / 16, 128, 0, stream>>>(x, basisA, basisB, (float*)d_out);
}
